// approx_Conv2d_int8_22900765622865
// MI455X (gfx1250) — 8-bit matrix operand order repaired and checked by reading, compiled, not run on hardware
//
#include <hip/hip_runtime.h>
#include <hip/hip_bf16.h>

typedef __attribute__((ext_vector_type(8))) int v8i;

#define HW     56
#define CIN    64
#define COUT   64
#define BATCH  8
#define NPOS   (HW * HW)
#define NTILES (NPOS / 16)
#define NELEM  (BATCH * CIN * NPOS)
#define WELEM  (COUT * CIN * 9)

#define WS_APACK_OFF 256
#define WS_XQ_OFF    37120

__global__ void k_init(unsigned int* s) {
    if (threadIdx.x < 2) s[threadIdx.x] = 0u;
}

__global__ void __launch_bounds__(256)
k_absmax(const float* __restrict__ src, int n, unsigned int* dst) {
    __shared__ float red[256];
    float m = 0.0f;
    for (int i = blockIdx.x * blockDim.x + threadIdx.x; i < n;
         i += gridDim.x * blockDim.x)
        m = fmaxf(m, fabsf(src[i]));
    red[threadIdx.x] = m;
    __syncthreads();
    for (int s = 128; s > 0; s >>= 1) {
        if (threadIdx.x < (unsigned)s)
            red[threadIdx.x] = fmaxf(red[threadIdx.x], red[threadIdx.x + s]);
        __syncthreads();
    }
    if (threadIdx.x == 0)
        atomicMax(dst, __float_as_uint(red[0]));
}

__global__ void __launch_bounds__(256)
k_quant_x(const float* __restrict__ x, const unsigned int* __restrict__ scales,
          signed char* __restrict__ xq) {
    int i = blockIdx.x * blockDim.x + threadIdx.x;
    if (i >= NELEM) return;
    float inv = 127.0f / __uint_as_float(scales[0]);
    int w = i % HW;  int t = i / HW;
    int h = t % HW;  t /= HW;
    int c = t % CIN; int b = t / CIN;
    float q = __builtin_rintf(x[i] * inv);
    q = fminf(fmaxf(q, -128.0f), 127.0f);
    xq[((b * HW + h) * HW + w) * CIN + c] = (signed char)(int)q;
}

__global__ void __launch_bounds__(256)
k_pack_w(const float* __restrict__ wgt, const unsigned int* __restrict__ scales,
         unsigned int* __restrict__ apack) {
    int i = blockIdx.x * blockDim.x + threadIdx.x;
    if (i >= 4 * 9 * 32 * 8) return;
    float inv = 127.0f / __uint_as_float(scales[1]);
    int v = i & 7;        int t = i >> 3;
    int lane = t & 31;    t >>= 5;
    int ktile = t % 9;    int mtile = t / 9;
    int half = lane >> 4;
    int m = mtile * 16 + (lane & 15);
    int kh = ktile / 3, kw = ktile % 3;
    int kbase = ((v >> 2) << 5) + (half << 4) + ((v & 3) << 2);
    unsigned int packed = 0;
    #pragma unroll
    for (int j = 0; j < 4; ++j) {
        int c = kbase + j;
        float q = __builtin_rintf(wgt[((m * CIN + c) * 3 + kh) * 3 + kw] * inv);
        q = fminf(fmaxf(q, -128.0f), 127.0f);
        packed |= ((unsigned int)((int)q & 0xFF)) << (8 * j);
    }
    apack[((mtile * 9 + ktile) * 32 + lane) * 8 + v] = packed;
}

__global__ void __launch_bounds__(256)
k_gemm(const signed char* __restrict__ xq,
       const unsigned int* __restrict__ apack,
       const unsigned int* __restrict__ scales,
       const float* __restrict__ bias,
       float* __restrict__ out) {
    int lane   = threadIdx.x & 31;
    int waveId = blockIdx.x * 8 + (threadIdx.x >> 5);
    int mtile  = waveId & 3;
    int rest   = waveId >> 2;
    int ntile  = rest % NTILES;
    int b      = rest / NTILES;

    int half = lane >> 4;
    int col  = lane & 15;
    int p  = ntile * 16 + col;
    int oh = p / HW, ow = p % HW;

    v8i acc = {0, 0, 0, 0, 0, 0, 0, 0};
    #pragma unroll
    for (int ktile = 0; ktile < 9; ++ktile) {
        const int4* al =
            (const int4*)(apack + ((size_t)((mtile * 9 + ktile) * 32 + lane)) * 8);
        int4 a0 = al[0], a1 = al[1];
        v8i A;
        A[0] = a0.x; A[1] = a0.y; A[2] = a0.z; A[3] = a0.w;
        A[4] = a1.x; A[5] = a1.y; A[6] = a1.z; A[7] = a1.w;

        int kh = ktile / 3, kw = ktile % 3;
        int h = oh + kh - 1, w = ow + kw - 1;
        int4 b0 = {0, 0, 0, 0}, b1 = {0, 0, 0, 0};
        if ((unsigned)h < (unsigned)HW && (unsigned)w < (unsigned)HW) {
            const int4* xr = (const int4*)(xq +
                (size_t)(((b * HW + h) * HW + w)) * CIN + half * 16);
            b0 = xr[0];
            b1 = xr[2];
        }
        v8i Bm;
        Bm[0] = b0.x; Bm[1] = b0.y; Bm[2] = b0.z; Bm[3] = b0.w;
        Bm[4] = b1.x; Bm[5] = b1.y; Bm[6] = b1.z; Bm[7] = b1.w;

        acc = __builtin_amdgcn_wmma_i32_16x16x64_iu8(
             true, A,  true, Bm, acc,
             false,  false);
    }

    float scale = (__uint_as_float(scales[0]) * (1.0f / 127.0f)) *
                  (__uint_as_float(scales[1]) * (1.0f / 127.0f));
    #pragma unroll
    for (int r = 0; r < 8; ++r) {
        int mg = mtile * 16 + r + half * 8;
        out[(((size_t)b * COUT + mg) * HW + oh) * HW + ow] =
            (float)acc[r] * scale + bias[mg];
    }
}

extern "C" void kernel_launch(void* const* d_in, const int* in_sizes, int n_in,
                              void* d_out, int out_size, void* d_ws, size_t ws_size,
                              hipStream_t stream) {
    const float* x    = (const float*)d_in[0];
    const float* wgt  = (const float*)d_in[1];
    const float* bias = (const float*)d_in[2];

    unsigned char* ws     = (unsigned char*)d_ws;
    unsigned int*  scales = (unsigned int*)ws;
    unsigned int*  apack  = (unsigned int*)(ws + WS_APACK_OFF);
    signed char*   xq     = (signed char*)(ws + WS_XQ_OFF);
    float*         out    = (float*)d_out;

    k_init<<<1, 32, 0, stream>>>(scales);
    k_absmax<<<512, 256, 0, stream>>>(x, NELEM, scales + 0);
    k_absmax<<<64, 256, 0, stream>>>(wgt, WELEM, scales + 1);
    k_quant_x<<<(NELEM + 255) / 256, 256, 0, stream>>>(x, scales, xq);
    k_pack_w<<<(4 * 9 * 32 * 8 + 255) / 256, 256, 0, stream>>>(wgt, scales, apack);
    k_gemm<<<784, 256, 0, stream>>>(xq, apack, scales, bias, out);
}
